// DecoderRNN_56289841381704
// MI455X (gfx1250) — matrix operand order repaired and checked by reading, not compiled, not run on hardware
//
#include <hip/hip_runtime.h>

// ---------------- problem constants ----------------
constexpr int NB = 32;      // batch
constexpr int NT = 256;     // decode steps
constexpr int NS = 512;     // encoder length
constexpr int NH = 512;     // hidden
constexpr int H3 = 3 * NH;  // gate width
constexpr int BT = NB * NT; // 8192

constexpr int LDK = 36;     // padded LDS k-stride for tiled GEMMs (bf16 elems)
constexpr int LDW = 520;    // padded LDS k-stride for persistent GRU (512 + 8)

typedef __attribute__((ext_vector_type(16))) __bf16    v16bf;
typedef __attribute__((ext_vector_type(8)))  float     v8f;
typedef __attribute__((ext_vector_type(4)))  unsigned  u32x4;
typedef __attribute__((ext_vector_type(8)))  unsigned  u32x8;

// ---------------- WMMA helpers ----------------
__device__ inline v8f wmma_bf16(v16bf a, v16bf b, v8f c) {
  return __builtin_amdgcn_wmma_f32_16x16x32_bf16(false, a, false, b,
                                                 (short)0, c, false, false);
}

// fp32 -> (hi,lo) bf16 pair; hi*hi + hi*lo + lo*hi ~= fp32 product
__device__ inline void split_bf16(float x, __bf16 &hi, __bf16 &lo) {
  hi = (__bf16)x;
  lo = (__bf16)(x - (float)hi);
}

// A fragment: 16x32 (MxK) tile in LDS, row-major, leading dim ld.
__device__ inline v16bf load_a_frag(const __bf16 *a, int ld, int lane) {
  const int half = lane >> 4, m = lane & 15;
  v16bf f;
#pragma unroll
  for (int e = 0; e < 16; ++e) {
    const int k = ((e & 8) << 1) + (((e >> 1) & 3) << 1) + (e & 1) + half * 8;
    f[e] = a[m * ld + k];
  }
  return f;
}

// B fragment: 32x16 (KxN) staged in LDS as [N][K] row-major, leading dim ld.
__device__ inline v16bf load_b_frag(const __bf16 *b, int ld, int lane) {
  const int half = lane >> 4, n = lane & 15;
  v16bf f;
#pragma unroll
  for (int e = 0; e < 16; ++e)
    f[e] = b[n * ld + ((e & 8) << 1) + half * 8 + (e & 7)];
  return f;
}

__device__ inline float sigmoidf_(float x) { return 1.0f / (1.0f + __expf(-x)); }

// ---------------- Tensor Data Mover (async tensor copy) ----------------
// D# group 0: count=1 | lds_addr | global_addr | type=2 (ISA 8.3)
__device__ inline u32x4 tdm_g0(const void *gptr, const void *lptr) {
  const unsigned long long ga = (unsigned long long)gptr;
  const unsigned lds = (unsigned)(unsigned long long)lptr;  // low 32 = LDS offset
  u32x4 g;
  g[0] = 1u;                                                // count=1, user mode
  g[1] = lds;                                               // lds_addr
  g[2] = (unsigned)ga;                                      // global_addr[31:0]
  g[3] = (unsigned)((ga >> 32) & 0x01FFFFFFu) | (2u << 30); // ga[56:32] | type=2
  return g;
}

// D# group 1 for a 32x512 bf16 tile, row padded +8 elems in LDS (ISA 8.4):
// data_size=2B, pad_enable, pad_interval=256 DW (one row), pad_amount=4 DW.
__device__ inline u32x8 tdm_g1_h() {
  u32x8 g;
  g[0] = (1u << 16) | (1u << 20) | (7u << 22) | (3u << 25);
  g[1] = 512u << 16;           // tensor_dim0[15:0] @ bits 63:48
  g[2] = (32u << 16);          // tensor_dim0 hi=0 | tensor_dim1=32
  g[3] = (512u << 16);         // tensor_dim1 hi=0 | tile_dim0=512
  g[4] = 32u;                  // tile_dim1=32, tile_dim2=0
  g[5] = 512u;                 // tensor_dim0_stride[31:0]
  g[6] = 0u;                   // stride hi | dim1_stride (unused, 2-D)
  g[7] = 0u;
  return g;
}

__device__ inline void tdm_load_2d(u32x4 g0, u32x8 g1) {
  asm volatile("tensor_load_to_lds %0, %1" :: "s"(g0), "s"(g1) : "memory");
}

// ======================================================================
// Kernel 1: xp = X(BT,NH) @ W_ih^T + b_ih  -> XP(BT,H3)
// ======================================================================
__global__ __launch_bounds__(256)
void xp_gemm_kernel(const float *__restrict__ X, const float *__restrict__ Wih,
                    const float *__restrict__ bih, float *__restrict__ XP) {
  __shared__ __bf16 Ah[64 * LDK], Al[64 * LDK], Bh[64 * LDK], Bl[64 * LDK];
  const int tid = threadIdx.x, wave = tid >> 5, lane = tid & 31;
  const int row0 = blockIdx.x * 64, col0 = blockIdx.y * 64;
  const int rt = wave & 3, ct0 = (wave >> 2) << 1;
  v8f acc0 = {}, acc1 = {};
  for (int kk = 0; kk < NH; kk += 32) {
    for (int i = tid; i < 64 * 32; i += 256) {
      const int r = i >> 5, c = i & 31;
      split_bf16(X[(size_t)(row0 + r) * NH + kk + c], Ah[r * LDK + c], Al[r * LDK + c]);
      split_bf16(Wih[(size_t)(col0 + r) * NH + kk + c], Bh[r * LDK + c], Bl[r * LDK + c]);
    }
    __syncthreads();
    const v16bf ah = load_a_frag(&Ah[rt * 16 * LDK], LDK, lane);
    const v16bf al = load_a_frag(&Al[rt * 16 * LDK], LDK, lane);
#pragma unroll
    for (int j = 0; j < 2; ++j) {
      const v16bf bh = load_b_frag(&Bh[(ct0 + j) * 16 * LDK], LDK, lane);
      const v16bf bl = load_b_frag(&Bl[(ct0 + j) * 16 * LDK], LDK, lane);
      v8f &acc = j ? acc1 : acc0;
      acc = wmma_bf16(ah, bh, acc);
      acc = wmma_bf16(ah, bl, acc);
      acc = wmma_bf16(al, bh, acc);
    }
    __syncthreads();
  }
  const int half = lane >> 4, nc = lane & 15;
#pragma unroll
  for (int e = 0; e < 8; ++e) {
    const int gr = row0 + rt * 16 + e + half * 8;
    const int c0 = col0 + ct0 * 16 + nc;
    XP[(size_t)gr * H3 + c0]      = acc0[e] + bih[c0];
    XP[(size_t)gr * H3 + c0 + 16] = acc1[e] + bih[c0 + 16];
  }
}

// ======================================================================
// Kernel 2: persistent GRU. 16 WGs x 128 threads, W_hh slice resident in
// LDS for all 256 steps; h staged per step via TDM (tensor_load_to_lds);
// software grid barrier between steps.
// LDS: Wh/Wl [3*32][LDW] + Hh/Hl [32][LDW]  (~260 KB dynamic)
// ======================================================================
__global__ __launch_bounds__(128)
void gru_persistent_kernel(const float *__restrict__ XP,
                           const float *__restrict__ Whh,
                           const float *__restrict__ bhh,
                           __bf16 *__restrict__ Gh,   // [2][NB*NH] h hi planes
                           __bf16 *__restrict__ Gl,   // [2][NB*NH] h lo planes
                           float *__restrict__ Yout,  // (B,T,H)
                           unsigned *__restrict__ cnt) {
  extern __shared__ __bf16 smem[];
  __bf16 *Wh = smem;                       // [3*32][LDW]
  __bf16 *Wl = Wh + 3 * 32 * LDW;
  __bf16 *Hh = Wl + 3 * 32 * LDW;          // [32][LDW]
  __bf16 *Hl = Hh + 32 * LDW;

  const int tid = threadIdx.x, wave = tid >> 5, lane = tid & 31;
  const int j0 = blockIdx.x * 32;          // this WG's 32 hidden units
  const int rt = wave & 1, jt = wave >> 1; // row tile (batch) / j tile

  // one-time: stage this WG's W_hh slice (split bf16) into LDS
  for (int i = tid; i < 3 * 32 * NH; i += 128) {
    const int k = i & (NH - 1);
    const int n = (i >> 9) & 31;
    const int g = i >> 14;
    const float v = Whh[(size_t)(g * NH + j0 + n) * NH + k];
    __bf16 hi = (__bf16)v;
    Wh[(g * 32 + n) * LDW + k] = hi;
    Wl[(g * 32 + n) * LDW + k] = (__bf16)(v - (float)hi);
  }

  const int half = lane >> 4, nc = lane & 15;
  const int j = j0 + jt * 16 + nc;         // this lane's hidden unit
  const float br = bhh[j], bz = bhh[NH + j], bn = bhh[2 * NH + j];

  // TDM descriptors for the two h parity planes
  const u32x8 g1 = tdm_g1_h();
  u32x4 g0h[2], g0l[2];
  g0h[0] = tdm_g0(Gh, Hh);            g0h[1] = tdm_g0(Gh + NB * NH, Hh);
  g0l[0] = tdm_g0(Gl, Hl);            g0l[1] = tdm_g0(Gl + NB * NH, Hl);

  __syncthreads();  // weights staged

  for (int t = 0; t < NT; ++t) {
    // async tensor copy of h(t) hi/lo planes into padded LDS
    if (wave == 0) {
      tdm_load_2d(g0h[t & 1], g1);
      tdm_load_2d(g0l[t & 1], g1);
      __builtin_amdgcn_s_wait_tensorcnt(0);
    }
    __syncthreads();

    v8f acc0 = {}, acc1 = {}, acc2 = {};
    const __bf16 *ah_base = Hh + rt * 16 * LDW;
    const __bf16 *al_base = Hl + rt * 16 * LDW;
    const __bf16 *bh_base = Wh + jt * 16 * LDW;
    const __bf16 *bl_base = Wl + jt * 16 * LDW;
#pragma unroll 4
    for (int kk = 0; kk < NH; kk += 32) {
      const v16bf ah = load_a_frag(ah_base + kk, LDW, lane);
      const v16bf al = load_a_frag(al_base + kk, LDW, lane);
#pragma unroll
      for (int g = 0; g < 3; ++g) {
        const v16bf bh = load_b_frag(bh_base + g * 32 * LDW + kk, LDW, lane);
        const v16bf bl = load_b_frag(bl_base + g * 32 * LDW + kk, LDW, lane);
        v8f &acc = (g == 0) ? acc0 : (g == 1) ? acc1 : acc2;
        acc = wmma_bf16(ah, bh, acc);
        acc = wmma_bf16(ah, bl, acc);
        acc = wmma_bf16(al, bh, acc);
      }
    }

    // fused gates + state update
    __bf16 *GhN = Gh + (size_t)((t + 1) & 1) * NB * NH;
    __bf16 *GlN = Gl + (size_t)((t + 1) & 1) * NB * NH;
#pragma unroll
    for (int e = 0; e < 8; ++e) {
      const int b = rt * 16 + e + half * 8;
      const size_t xrow = ((size_t)b * NT + t) * H3;
      const float r = sigmoidf_(XP[xrow + j] + acc0[e] + br);
      const float z = sigmoidf_(XP[xrow + NH + j] + acc1[e] + bz);
      const float n = tanhf(XP[xrow + 2 * NH + j] + r * (acc2[e] + bn));
      const float hp = (float)Hh[b * LDW + j] + (float)Hl[b * LDW + j];
      const float hn = (1.0f - z) * n + z * hp;
      const __bf16 hi = (__bf16)hn;
      GhN[b * NH + j] = hi;
      GlN[b * NH + j] = (__bf16)(hn - (float)hi);
      Yout[((size_t)b * NT + t) * NH + j] = hn;
      if (t + 1 < NT) __builtin_prefetch(&XP[xrow + H3 + j], 0, 1);
    }

    // grid barrier: release h(t+1), acquire everyone else's
    __syncthreads();
    if (tid == 0) {
      __hip_atomic_fetch_add(cnt, 1u, __ATOMIC_ACQ_REL, __HIP_MEMORY_SCOPE_AGENT);
      const unsigned target = 16u * (unsigned)(t + 1);
      while (__hip_atomic_load(cnt, __ATOMIC_ACQUIRE, __HIP_MEMORY_SCOPE_AGENT) < target)
        __builtin_amdgcn_s_sleep(1);
    }
    __syncthreads();
  }
}

// ======================================================================
// Kernel 3: scores = Y @ Enc[b]^T, row softmax, write attn in (T,B,S).
// ======================================================================
__global__ __launch_bounds__(256)
void attn_softmax_kernel(const float *__restrict__ Y, const float *__restrict__ Enc,
                         float *__restrict__ Attn) {
  __shared__ __bf16 Ah[16 * LDK], Al[16 * LDK], Bh[128 * LDK], Bl[128 * LDK];
  __shared__ float Scr[16 * NS];
  __shared__ float Red[16 * 16];
  const int tid = threadIdx.x, wave = tid >> 5, lane = tid & 31;
  const int b = blockIdx.x, t0 = blockIdx.y * 16;
  v8f acc[4] = {{}, {}, {}, {}};
#pragma unroll
  for (int cg = 0; cg < 4; ++cg) {
    const int col0 = cg * 128;
    for (int kk = 0; kk < NH; kk += 32) {
      for (int i = tid; i < 16 * 32; i += 256) {
        const int r = i >> 5, c = i & 31;
        split_bf16(Y[((size_t)b * NT + t0 + r) * NH + kk + c],
                   Ah[r * LDK + c], Al[r * LDK + c]);
      }
      for (int i = tid; i < 128 * 32; i += 256) {
        const int n = i >> 5, c = i & 31;
        split_bf16(Enc[((size_t)b * NS + col0 + n) * NH + kk + c],
                   Bh[n * LDK + c], Bl[n * LDK + c]);
      }
      __syncthreads();
      const v16bf ah = load_a_frag(Ah, LDK, lane);
      const v16bf al = load_a_frag(Al, LDK, lane);
      const v16bf bh = load_b_frag(&Bh[wave * 16 * LDK], LDK, lane);
      const v16bf bl = load_b_frag(&Bl[wave * 16 * LDK], LDK, lane);
      acc[cg] = wmma_bf16(ah, bh, acc[cg]);
      acc[cg] = wmma_bf16(ah, bl, acc[cg]);
      acc[cg] = wmma_bf16(al, bh, acc[cg]);
      __syncthreads();
    }
  }
  const int half = lane >> 4, nc = lane & 15;
#pragma unroll
  for (int cg = 0; cg < 4; ++cg)
#pragma unroll
    for (int e = 0; e < 8; ++e)
      Scr[(e + half * 8) * NS + cg * 128 + wave * 16 + nc] = acc[cg][e];
  __syncthreads();
  const int r = tid >> 4, cseg = (tid & 15) * 32;
  float mx = -3.4e38f;
  for (int i = 0; i < 32; ++i) mx = fmaxf(mx, Scr[r * NS + cseg + i]);
  Red[r * 16 + (tid & 15)] = mx;
  __syncthreads();
  mx = Red[r * 16];
  for (int i = 1; i < 16; ++i) mx = fmaxf(mx, Red[r * 16 + i]);
  float sum = 0.0f;
  for (int i = 0; i < 32; ++i) {
    const float e = __expf(Scr[r * NS + cseg + i] - mx);
    Scr[r * NS + cseg + i] = e;
    sum += e;
  }
  __syncthreads();
  Red[r * 16 + (tid & 15)] = sum;
  __syncthreads();
  sum = 0.0f;
  for (int i = 0; i < 16; ++i) sum += Red[r * 16 + i];
  const float inv = 1.0f / sum;
  for (int i = 0; i < 32; ++i)
    Attn[(((size_t)(t0 + r)) * NB + b) * NS + cseg + i] = Scr[r * NS + cseg + i] * inv;
}

// ======================================================================
// Kernel 4: mix[b] = attn[b](T,S) @ Enc[b](S,H); attn read from (T,B,S).
// ======================================================================
__global__ __launch_bounds__(256)
void mix_gemm_kernel(const float *__restrict__ Attn, const float *__restrict__ Enc,
                     float *__restrict__ Mix) {
  __shared__ __bf16 Ah[64 * LDK], Al[64 * LDK], Bh[64 * LDK], Bl[64 * LDK];
  const int tid = threadIdx.x, wave = tid >> 5, lane = tid & 31;
  const int row0 = blockIdx.x * 64, col0 = blockIdx.y * 64;
  const int b = row0 / NT, t0 = row0 % NT;
  const int rt = wave & 3, ct0 = (wave >> 2) << 1;
  v8f acc0 = {}, acc1 = {};
  for (int kk = 0; kk < NS; kk += 32) {
    for (int i = tid; i < 64 * 32; i += 256) {
      const int r = i >> 5, c = i & 31;
      split_bf16(Attn[(((size_t)(t0 + r)) * NB + b) * NS + kk + c],
                 Ah[r * LDK + c], Al[r * LDK + c]);
    }
    for (int i = tid; i < 32 * 64; i += 256) {
      const int k = i >> 6, n = i & 63;
      split_bf16(Enc[((size_t)b * NS + kk + k) * NH + col0 + n],
                 Bh[n * LDK + k], Bl[n * LDK + k]);
    }
    __syncthreads();
    const v16bf ah = load_a_frag(&Ah[rt * 16 * LDK], LDK, lane);
    const v16bf al = load_a_frag(&Al[rt * 16 * LDK], LDK, lane);
#pragma unroll
    for (int j = 0; j < 2; ++j) {
      const v16bf bh = load_b_frag(&Bh[(ct0 + j) * 16 * LDK], LDK, lane);
      const v16bf bl = load_b_frag(&Bl[(ct0 + j) * 16 * LDK], LDK, lane);
      v8f &acc = j ? acc1 : acc0;
      acc = wmma_bf16(ah, bh, acc);
      acc = wmma_bf16(ah, bl, acc);
      acc = wmma_bf16(al, bh, acc);
    }
    __syncthreads();
  }
  const int half = lane >> 4, nc = lane & 15;
#pragma unroll
  for (int e = 0; e < 8; ++e) {
    const int gr = row0 + rt * 16 + e + half * 8;
    const int c0 = col0 + ct0 * 16 + nc;
    Mix[(size_t)gr * NH + c0]      = acc0[e];
    Mix[(size_t)gr * NH + c0 + 16] = acc1[e];
  }
}

// ======================================================================
// Kernel 5: att_out = tanh(mix @ Wm^T + out @ Wo2^T + b_out) -> (T,B,H)
// ======================================================================
__global__ __launch_bounds__(256)
void out_proj_kernel(const float *__restrict__ Mix, const float *__restrict__ Y,
                     const float *__restrict__ Wout, const float *__restrict__ bout,
                     float *__restrict__ Dec) {
  __shared__ __bf16 Ah[64 * LDK], Al[64 * LDK], Bh[64 * LDK], Bl[64 * LDK];
  const int tid = threadIdx.x, wave = tid >> 5, lane = tid & 31;
  const int row0 = blockIdx.x * 64, col0 = blockIdx.y * 64;
  const int rt = wave & 3, ct0 = (wave >> 2) << 1;
  v8f acc0 = {}, acc1 = {};
#pragma unroll
  for (int pass = 0; pass < 2; ++pass) {
    const float *A = pass ? Y : Mix;
    const int koff = pass ? NH : 0;
    for (int kk = 0; kk < NH; kk += 32) {
      for (int i = tid; i < 64 * 32; i += 256) {
        const int r = i >> 5, c = i & 31;
        split_bf16(A[(size_t)(row0 + r) * NH + kk + c], Ah[r * LDK + c], Al[r * LDK + c]);
        split_bf16(Wout[(size_t)(col0 + r) * (2 * NH) + koff + kk + c],
                   Bh[r * LDK + c], Bl[r * LDK + c]);
      }
      __syncthreads();
      const v16bf ah = load_a_frag(&Ah[rt * 16 * LDK], LDK, lane);
      const v16bf al = load_a_frag(&Al[rt * 16 * LDK], LDK, lane);
#pragma unroll
      for (int j = 0; j < 2; ++j) {
        const v16bf bh = load_b_frag(&Bh[(ct0 + j) * 16 * LDK], LDK, lane);
        const v16bf bl = load_b_frag(&Bl[(ct0 + j) * 16 * LDK], LDK, lane);
        v8f &acc = j ? acc1 : acc0;
        acc = wmma_bf16(ah, bh, acc);
        acc = wmma_bf16(ah, bl, acc);
        acc = wmma_bf16(al, bh, acc);
      }
      __syncthreads();
    }
  }
  const int half = lane >> 4, nc = lane & 15;
  const int b = row0 / NT, t0 = row0 % NT;
#pragma unroll
  for (int e = 0; e < 8; ++e) {
    const int t = t0 + rt * 16 + e + half * 8;
    const int c0 = col0 + ct0 * 16 + nc;
    Dec[((size_t)t * NB + b) * NH + c0]      = tanhf(acc0[e] + bout[c0]);
    Dec[((size_t)t * NB + b) * NH + c0 + 16] = tanhf(acc1[e] + bout[c0 + 16]);
  }
}

// ---------------- small setup / teardown kernels ----------------
__global__ void init_kernel(const float *__restrict__ Eh, __bf16 *__restrict__ Gh0,
                            __bf16 *__restrict__ Gl0, unsigned *__restrict__ cnt) {
  const int i = blockIdx.x * blockDim.x + threadIdx.x;
  if (i == 0) *cnt = 0u;
  if (i < NB * NH) {
    const float v = Eh[i];
    const __bf16 hi = (__bf16)v;
    Gh0[i] = hi;
    Gl0[i] = (__bf16)(v - (float)hi);
  }
}

__global__ void hid_kernel(const __bf16 *__restrict__ Gh0, const __bf16 *__restrict__ Gl0,
                           float *__restrict__ dst) {
  const int i = blockIdx.x * blockDim.x + threadIdx.x;
  if (i < NB * NH) dst[i] = (float)Gh0[i] + (float)Gl0[i];
}

// ======================================================================
extern "C" void kernel_launch(void* const* d_in, const int* in_sizes, int n_in,
                              void* d_out, int out_size, void* d_ws, size_t ws_size,
                              hipStream_t stream) {
  const float *X    = (const float *)d_in[0]; // (B,T,H)
  const float *Eh   = (const float *)d_in[1]; // (1,B,H)
  const float *Enc  = (const float *)d_in[2]; // (B,S,H)
  const float *Wih  = (const float *)d_in[3]; // (3H,H)
  const float *Whh  = (const float *)d_in[4]; // (3H,H)
  const float *bih  = (const float *)d_in[5];
  const float *bhh  = (const float *)d_in[6];
  const float *Wout = (const float *)d_in[7]; // (H,2H)
  const float *bout = (const float *)d_in[8];

  float *out     = (float *)d_out;
  float *dec_out = out;                            // (T,B,H)
  float *dec_hid = out + (size_t)NT * NB * NH;     // (1,B,H)
  float *attn    = dec_hid + (size_t)NB * NH;      // (T,B,S)

  float  *xp  = (float *)d_ws;                     // (BT,3H)
  float  *y   = xp + (size_t)BT * H3;              // GRU outputs (B,T,H)
  float  *mix = y + (size_t)BT * NH;               // (BT,H)
  __bf16 *Gh  = (__bf16 *)(mix + (size_t)BT * NH); // [2][NB*NH] h hi planes
  __bf16 *Gl  = Gh + 2 * (size_t)NB * NH;          // [2][NB*NH] h lo planes
  unsigned *cnt = (unsigned *)(Gl + 2 * (size_t)NB * NH);

  // h0 split planes + barrier counter
  init_kernel<<<(NB * NH + 255) / 256, 256, 0, stream>>>(Eh, Gh, Gl, cnt);

  // input projection (one big GEMM)
  xp_gemm_kernel<<<dim3(BT / 64, H3 / 64), 256, 0, stream>>>(X, Wih, bih, xp);

  // persistent recurrence: weights resident in LDS, TDM-staged h, grid barrier
  const size_t lds_bytes = (size_t)(2 * 3 * 32 * LDW + 2 * 32 * LDW) * sizeof(__bf16);
  gru_persistent_kernel<<<16, 128, lds_bytes, stream>>>(xp, Whh, bhh, Gh, Gl, y, cnt);

  // attention: scores + softmax fused, attn written in (T,B,S)
  attn_softmax_kernel<<<dim3(NB, NT / 16), 256, 0, stream>>>(y, Enc, attn);

  // context vectors
  mix_gemm_kernel<<<dim3(BT / 64, NH / 64), 256, 0, stream>>>(attn, Enc, mix);

  // fused concat + projection + tanh, transposed store to (T,B,H)
  out_proj_kernel<<<dim3(BT / 64, NH / 64), 256, 0, stream>>>(mix, y, Wout, bout, dec_out);

  // final hidden (parity 0 after 256 steps)
  hid_kernel<<<(NB * NH + 255) / 256, 256, 0, stream>>>(Gh, Gl, dec_hid);
}
